// MultiHeadAttention_63024350102323
// MI455X (gfx1250) — matrix operand order repaired and checked by reading, not compiled, not run on hardware
//
#include <hip/hip_runtime.h>
#include <hip/hip_bf16.h>

#define IN_CH   128
#define OUT_CH  32
#define HEADS   8
#define C_TOT   256            // HEADS*OUT_CH
#define NEG_SLOPE 0.2f

typedef __attribute__((ext_vector_type(16))) __bf16 v16bf;
typedef __attribute__((ext_vector_type(8)))  __bf16 v8bf;
typedef __attribute__((ext_vector_type(8)))  float  v8f;

// ---------- helpers ----------
__device__ __forceinline__ unsigned short f2bf(float f) {
  unsigned u = __float_as_uint(f);
  u += 0x7FFFu + ((u >> 16) & 1u);          // round-to-nearest-even
  return (unsigned short)(u >> 16);
}
// monotonic float <-> u32 encoding for atomicMax on floats
__device__ __forceinline__ unsigned fenc(float f) {
  unsigned u = __float_as_uint(f);
  return u ^ ((u >> 31) ? 0xFFFFFFFFu : 0x80000000u);
}
__device__ __forceinline__ float fdec(unsigned u) {
  unsigned v = u ^ ((u >> 31) ? 0x80000000u : 0xFFFFFFFFu);
  return __uint_as_float(v);
}
__device__ __forceinline__ float lrelu(float v) { return v > 0.f ? v : NEG_SLOPE * v; }

// ---------- 1) fp32 -> bf16 conversion of x ----------
__global__ void k_convert_x(const float* __restrict__ x, unsigned short* __restrict__ xb,
                            int n_elem) {
  int i = (blockIdx.x * blockDim.x + threadIdx.x) * 4;
  if (i + 3 < n_elem) {
    float4 f = *(const float4*)(x + i);
    unsigned lo = (unsigned)f2bf(f.x) | ((unsigned)f2bf(f.y) << 16);
    unsigned hi = (unsigned)f2bf(f.z) | ((unsigned)f2bf(f.w) << 16);
    *(uint2*)(xb + i) = uint2{lo, hi};
  } else {
    for (; i < n_elem; ++i) xb[i] = f2bf(x[i]);
  }
}

// ---------- 2) build Bt[c][k] = bf16(W[h][k][o]), c = h*32+o (K-contiguous) ----------
__global__ void k_build_bt(const float* __restrict__ W, unsigned short* __restrict__ Bt) {
  int idx = blockIdx.x * blockDim.x + threadIdx.x;     // c*128 + k
  if (idx >= C_TOT * IN_CH) return;
  int c = idx >> 7, k = idx & 127;
  int h = c >> 5, o = c & 31;
  Bt[idx] = f2bf(W[((size_t)h * IN_CH + k) * OUT_CH + o]);
}

// ---------- 3) WMMA GEMM: xw[N][256] = x[N][128] * B[128][256] ----------
__global__ __launch_bounds__(256) void k_gemm(const unsigned short* __restrict__ xb,
                                              const unsigned short* __restrict__ Btus,
                                              float* __restrict__ xw) {
  __shared__ __bf16 sA[16 * IN_CH];                    // 4 KB A tile
  const int mt = blockIdx.x;
  const int t  = threadIdx.x;

  // prefetch next M-tile of x into cache (speculative, OOB-safe)
  __builtin_prefetch(xb + ((size_t)mt + 1) * 16 * IN_CH + (size_t)t * 8, 0, 3);

  // stage A tile: 16 rows x 128 bf16 = 4096 B; one uint4 per thread
  ((uint4*)sA)[t] = ((const uint4*)(xb + (size_t)mt * 16 * IN_CH))[t];
  __syncthreads();

  const int w = t >> 5, l = t & 31;
  const int m = l & 15, hh = l >> 4;                   // row in tile, lane half

  // A fragments, contiguous K order (matches B): K = kk*32 + hh*16 + e
  v16bf afr[4];
#pragma unroll
  for (int kk = 0; kk < 4; ++kk) {
    union { v16bf v; v8bf h2[2]; } av;
    const __bf16* base = &sA[m * IN_CH + kk * 32 + hh * 16];
    av.h2[0] = *(const v8bf*)(base);
    av.h2[1] = *(const v8bf*)(base + 8);
    afr[kk] = av.v;
  }

  const __bf16* Bt = (const __bf16*)Btus;
#pragma unroll
  for (int c2 = 0; c2 < 2; ++c2) {
    const int ct = w * 2 + c2;                         // column tile 0..15
    v8f acc = {};
#pragma unroll
    for (int kk = 0; kk < 4; ++kk) {
      // B column n, 16 contiguous K values per lane half (32B aligned load)
      const int n = ct * 16 + m;
      v16bf bfr = *(const v16bf*)(Bt + (size_t)n * IN_CH + kk * 32 + hh * 16);
      acc = __builtin_amdgcn_wmma_f32_16x16x32_bf16(
          false, afr[kk], false, bfr, (short)0, acc, false, false);
    }
    // C layout: lane half hh -> rows hh*8 + r, column = ct*16 + m
    float* dst = xw + ((size_t)mt * 16 + hh * 8) * C_TOT + ct * 16 + m;
#pragma unroll
    for (int r = 0; r < 8; ++r) dst[(size_t)r * C_TOT] = acc[r];
  }
}

// ---------- 4) attention scores a_s / a_d ----------
__global__ void k_attn(const float* __restrict__ xw, const float* __restrict__ att_src,
                       const float* __restrict__ att_dst, float* __restrict__ a_s,
                       float* __restrict__ a_d, int NH) {
  int i = blockIdx.x * blockDim.x + threadIdx.x;
  if (i >= NH) return;
  int n = i >> 3, h = i & 7;
  const float* row = xw + (size_t)n * C_TOT + h * OUT_CH;
  const float* as  = att_src + h * OUT_CH;
  const float* ad  = att_dst + h * OUT_CH;
  float s = 0.f, d = 0.f;
#pragma unroll
  for (int o = 0; o < OUT_CH; ++o) { float v = row[o]; s += v * as[o]; d += v * ad[o]; }
  a_s[i] = s;  a_d[i] = d;
}

// ---------- 5) init: out = bias, mbits = 0 (== -inf), denom = 0 ----------
__global__ void k_init(float* __restrict__ out, const float* __restrict__ bias,
                       unsigned* __restrict__ mbits, float* __restrict__ denom,
                       int NC, int NH) {
  int i = blockIdx.x * blockDim.x + threadIdx.x;
  if (i < NC) out[i] = bias[i & (C_TOT - 1)];
  if (i < NH) { mbits[i] = 0u; denom[i] = 0.f; }
}

// ---------- 6) edge pass 1: segment max ----------
__global__ void k_edge_max(const int* __restrict__ ei, int E, int Etot,
                           const float* __restrict__ a_s, const float* __restrict__ a_d,
                           unsigned* __restrict__ mbits) {
  int i = blockIdx.x * blockDim.x + threadIdx.x;
  if (i >= Etot * HEADS) return;
  int e = i >> 3, h = i & 7;
  int s, d;
  if (e < E) { s = ei[e]; d = ei[E + e]; } else { s = d = e - E; }
  float v = lrelu(a_s[s * HEADS + h] + a_d[d * HEADS + h]);
  atomicMax(&mbits[d * HEADS + h], fenc(v));
}

// ---------- 7) edge pass 2: segment sum of exp ----------
__global__ void k_edge_sum(const int* __restrict__ ei, int E, int Etot,
                           const float* __restrict__ a_s, const float* __restrict__ a_d,
                           const unsigned* __restrict__ mbits, float* __restrict__ denom) {
  int i = blockIdx.x * blockDim.x + threadIdx.x;
  if (i >= Etot * HEADS) return;
  int e = i >> 3, h = i & 7;
  int s, d;
  if (e < E) { s = ei[e]; d = ei[E + e]; } else { s = d = e - E; }
  float v = lrelu(a_s[s * HEADS + h] + a_d[d * HEADS + h]);
  float p = __expf(v - fdec(mbits[d * HEADS + h]));
  atomicAdd(&denom[d * HEADS + h], p);
}

// ---------- 8) edge pass 3: alpha-weighted scatter-add of xw rows ----------
__global__ void k_edge_scatter(const int* __restrict__ ei, int E, int Etot,
                               const float* __restrict__ a_s, const float* __restrict__ a_d,
                               const unsigned* __restrict__ mbits,
                               const float* __restrict__ denom,
                               const float* __restrict__ xw, float* __restrict__ out) {
  int i = blockIdx.x * blockDim.x + threadIdx.x;
  if (i >= Etot * HEADS) return;
  int e = i >> 3, h = i & 7;
  int s, d;
  if (e < E) { s = ei[e]; d = ei[E + e]; } else { s = d = e - E; }
  float v = lrelu(a_s[s * HEADS + h] + a_d[d * HEADS + h]);
  float alpha = __expf(v - fdec(mbits[d * HEADS + h])) / denom[d * HEADS + h];
  const float4* src = (const float4*)(xw + (size_t)s * C_TOT + h * OUT_CH);
  float* dst = out + (size_t)d * C_TOT + h * OUT_CH;
#pragma unroll
  for (int q = 0; q < 8; ++q) {
    float4 f = src[q];
    atomicAdd(dst + q * 4 + 0, alpha * f.x);
    atomicAdd(dst + q * 4 + 1, alpha * f.y);
    atomicAdd(dst + q * 4 + 2, alpha * f.z);
    atomicAdd(dst + q * 4 + 3, alpha * f.w);
  }
}

extern "C" void kernel_launch(void* const* d_in, const int* in_sizes, int n_in,
                              void* d_out, int out_size, void* d_ws, size_t ws_size,
                              hipStream_t stream) {
  const float* x       = (const float*)d_in[0];
  const int*   ei      = (const int*)d_in[1];
  const float* W       = (const float*)d_in[2];
  const float* att_src = (const float*)d_in[3];
  const float* att_dst = (const float*)d_in[4];
  const float* bias    = (const float*)d_in[5];
  float* out = (float*)d_out;

  const int N    = in_sizes[0] / IN_CH;
  const int E    = in_sizes[1] / 2;
  const int Etot = E + N;

  // workspace carve-out (256B aligned)
  char* ws = (char*)d_ws;
  size_t off = 0;
  auto carve = [&](size_t bytes) -> void* {
    off = (off + 255) & ~(size_t)255;
    void* p = ws + off;
    off += bytes;
    return p;
  };
  unsigned short* xb    = (unsigned short*)carve((size_t)N * IN_CH * 2);
  unsigned short* Bt    = (unsigned short*)carve((size_t)C_TOT * IN_CH * 2);
  float*          xw    = (float*)carve((size_t)N * C_TOT * 4);
  float*          a_s   = (float*)carve((size_t)N * HEADS * 4);
  float*          a_d   = (float*)carve((size_t)N * HEADS * 4);
  unsigned*       mbits = (unsigned*)carve((size_t)N * HEADS * 4);
  float*          denom = (float*)carve((size_t)N * HEADS * 4);
  (void)ws_size; (void)n_in; (void)out_size;

  const int TB = 256;
  // 1) convert x to bf16
  int nx = N * IN_CH;
  k_convert_x<<<(nx / 4 + TB - 1) / TB, TB, 0, stream>>>(x, xb, nx);
  // 2) build transposed bf16 weight matrix
  k_build_bt<<<(C_TOT * IN_CH + TB - 1) / TB, TB, 0, stream>>>(W, Bt);
  // 3) WMMA GEMM (one 16-row tile per block, 8 waves x 2 column tiles)
  k_gemm<<<N / 16, TB, 0, stream>>>(xb, Bt, xw);
  // 4) per-node attention scores
  k_attn<<<(N * HEADS + TB - 1) / TB, TB, 0, stream>>>(xw, att_src, att_dst, a_s, a_d,
                                                       N * HEADS);
  // 5) init out/max/denom
  k_init<<<(N * C_TOT + TB - 1) / TB, TB, 0, stream>>>(out, bias, mbits, denom,
                                                       N * C_TOT, N * HEADS);
  // 6-8) edge phase
  int ne = Etot * HEADS;
  k_edge_max<<<(ne + TB - 1) / TB, TB, 0, stream>>>(ei, E, Etot, a_s, a_d, mbits);
  k_edge_sum<<<(ne + TB - 1) / TB, TB, 0, stream>>>(ei, E, Etot, a_s, a_d, mbits, denom);
  k_edge_scatter<<<(ne + TB - 1) / TB, TB, 0, stream>>>(ei, E, Etot, a_s, a_d, mbits,
                                                        denom, xw, out);
}
